// GeometricLSSViewTransformer_403726926244
// MI455X (gfx1250) — matrix operand order repaired and checked by reading, not compiled, not run on hardware
//
#include <hip/hip_runtime.h>
#include <hip/hip_bf16.h>
#include <math.h>

// ---------------- problem constants ----------------
#define BN_   12          // B*N
#define B_    2
#define CIN_  256
#define COUT_ 80
#define D_    16
#define MOUT_ 96          // D + C_OUT
#define HF_   32
#define WF_   88
#define HW_   2816        // HF*WF
#define NB_   88          // HW / 32  (32 columns per block)
#define BEV_  128
#define CPAD_ 96          // C_OUT padded to 3*32 for WMMA K
#define BPAD_ 130         // 128 + halo

typedef __attribute__((ext_vector_type(16))) __bf16 v16bf;
typedef __attribute__((ext_vector_type(8)))  __bf16 v8bf;
typedef __attribute__((ext_vector_type(8)))  float  v8f;

// ---------------- workspace layout (bytes) ----------------
static const size_t OFF_Y      = 0;                                              // (BN,96,HW) f32
static const size_t OFF_DEPTH  = OFF_Y      + (size_t)BN_*MOUT_*HW_*4;           // (BN,HW,16) f32
static const size_t OFF_FLAT   = OFF_DEPTH  + (size_t)BN_*HW_*D_*4;              // (BN,HW,16) i32
static const size_t OFF_PARAMS = OFF_FLAT   + (size_t)BN_*HW_*D_*4;              // (BN,24)    f32
static const size_t OFF_ZERO   = (OFF_PARAMS + BN_*24*4 + 255) & ~(size_t)255;
static const size_t OFF_CNT    = OFF_ZERO;                                       // (B,128,128) f32
static const size_t OFF_ACC    = OFF_CNT    + (size_t)B_*BEV_*BEV_*4;            // (cells,80)  f32
static const size_t OFF_BEV    = OFF_ACC    + (size_t)B_*BEV_*BEV_*COUT_*4;      // (B,130,130,96) bf16
static const size_t OFF_WTAP   = OFF_BEV    + (size_t)B_*BPAD_*BPAD_*CPAD_*2;    // (9,80,96) bf16
static const size_t OFF_END    = OFF_WTAP   + (size_t)9*COUT_*CPAD_*2;

// =====================================================================
// Kernel 1: depthnet 1x1 conv as bf16 WMMA GEMM   Y = W(96x256) @ X(256xHW)
// block = 192 threads (6 waves, one 16-row tile each, 32 columns/block)
// grid = BN * 88.  A fragment reused across two N-subtiles (2 WMMA / chunk).
// =====================================================================
__global__ __launch_bounds__(192) void k_depth_gemm(
    const float* __restrict__ x, const float* __restrict__ w_dep,
    const float* __restrict__ b_dep, float* __restrict__ Y)
{
  __shared__ __bf16 bt[32 * 40];           // [n][k] pitch 40 halfs (bank-friendly)
  const int bn   = blockIdx.x / NB_;
  const int n0   = (blockIdx.x % NB_) * 32;
  const int tid  = threadIdx.x;
  const int wave = tid >> 5;               // m-tile 0..5
  const int lane = tid & 31;
  const int hn   = lane & 15;
  const int half = lane >> 4;

  const float* xb = x + (size_t)bn * CIN_ * HW_ + n0;
  v8f c0 = {}, c1 = {};
  for (int kc = 0; kc < 8; ++kc) {
    __syncthreads();
    // stage 32(K) x 32(N) fp32 tile of X -> bf16 LDS transposed to [n][k].
    // 256 float4 segments; straight-line so global side is pure b128 loads.
    {
      int e = tid;
      if (e < 256) {
        const int kk = e >> 3, n4 = (e & 7) * 4;
        const float4 f = *(const float4*)&xb[(size_t)(kc * 32 + kk) * HW_ + n4];
        bt[(n4 + 0) * 40 + kk] = (__bf16)f.x;
        bt[(n4 + 1) * 40 + kk] = (__bf16)f.y;
        bt[(n4 + 2) * 40 + kk] = (__bf16)f.z;
        bt[(n4 + 3) * 40 + kk] = (__bf16)f.w;
      }
      e = tid + 192;
      if (e < 256) {
        const int kk = e >> 3, n4 = (e & 7) * 4;
        const float4 f = *(const float4*)&xb[(size_t)(kc * 32 + kk) * HW_ + n4];
        bt[(n4 + 0) * 40 + kk] = (__bf16)f.x;
        bt[(n4 + 1) * 40 + kk] = (__bf16)f.y;
        bt[(n4 + 2) * 40 + kk] = (__bf16)f.z;
        bt[(n4 + 3) * 40 + kk] = (__bf16)f.w;
      }
    }
    if (kc < 7)  // warm L2/WGP$ for the next K-chunk
      __builtin_prefetch(xb + (size_t)((kc + 1) * 32) * HW_, 0, 1);
    __syncthreads();

    // A fragment: lane<16 holds K 0..15, lane>=16 holds K 16..31
    const int m = wave * 16 + hn;
    const float* wrow = w_dep + m * CIN_ + kc * 32 + half * 16;
    v16bf a, b0, b1;
    #pragma unroll
    for (int j = 0; j < 8; ++j) {
      a[j]     = (__bf16)wrow[j];
      a[j + 8] = (__bf16)wrow[j + 8];
    }
    // B fragments: lane n holds column n, K 0..15 (half=0) / 16..31 (half=1)
    {
      const v8bf lo = *(const v8bf*)&bt[hn * 40 + half * 16];
      const v8bf hi = *(const v8bf*)&bt[hn * 40 + half * 16 + 8];
      #pragma unroll
      for (int j = 0; j < 8; ++j) { b0[j] = lo[j]; b0[j + 8] = hi[j]; }
    }
    {
      const v8bf lo = *(const v8bf*)&bt[(hn + 16) * 40 + half * 16];
      const v8bf hi = *(const v8bf*)&bt[(hn + 16) * 40 + half * 16 + 8];
      #pragma unroll
      for (int j = 0; j < 8; ++j) { b1[j] = lo[j]; b1[j + 8] = hi[j]; }
    }
    c0 = __builtin_amdgcn_wmma_f32_16x16x32_bf16(false, a, false, b0,
                                                 (short)0, c0, false, false);
    c1 = __builtin_amdgcn_wmma_f32_16x16x32_bf16(false, a, false, b1,
                                                 (short)0, c1, false, false);
  }
  // D layout: c[v] -> row (wave*16 + v + half*8), col (n0 + hn [+16])
  const int col0 = n0 + hn;
  #pragma unroll
  for (int v = 0; v < 8; ++v) {
    const int row = wave * 16 + v + half * 8;
    const float bias = b_dep[row];
    float* yr = Y + ((size_t)bn * MOUT_ + row) * HW_;
    yr[col0]      = c0[v] + bias;
    yr[col0 + 16] = c1[v] + bias;
  }
}

// =====================================================================
// Kernel 2: per-camera matrix inverses (24 tiny 3x3 inverses, 2 per camera)
// =====================================================================
__device__ inline void inv3(const float* A, float* O) {
  const float a=A[0],b=A[1],cc=A[2],d=A[3],e=A[4],f=A[5],g=A[6],h=A[7],i=A[8];
  const float det = a*(e*i-f*h) - b*(d*i-f*g) + cc*(d*h-e*g);
  const float r = 1.0f / det;
  O[0]=(e*i-f*h)*r; O[1]=(cc*h-b*i)*r; O[2]=(b*f-cc*e)*r;
  O[3]=(f*g-d*i)*r; O[4]=(a*i-cc*g)*r; O[5]=(cc*d-a*f)*r;
  O[6]=(d*h-e*g)*r; O[7]=(b*g-a*h)*r;  O[8]=(a*e-b*d)*r;
}

__global__ void k_params(const float* __restrict__ l2i,
                         const float* __restrict__ aug,
                         float* __restrict__ P)
{
  const int bn = threadIdx.x;
  if (bn >= BN_) return;
  const float* A = aug + bn * 16;   // 4x4 row-major
  const float* M = l2i + bn * 16;
  float* p = P + bn * 24;
  float R[9];
  R[0]=A[0]; R[1]=A[1]; R[2]=A[2]; R[3]=A[4]; R[4]=A[5]; R[5]=A[6];
  R[6]=A[8]; R[7]=A[9]; R[8]=A[10];
  inv3(R, p);                       // invAug
  p[9]=A[3]; p[10]=A[7]; p[11]=A[11];   // tAug
  R[0]=M[0]; R[1]=M[1]; R[2]=M[2]; R[3]=M[4]; R[4]=M[5]; R[5]=M[6];
  R[6]=M[8]; R[7]=M[9]; R[8]=M[10];
  float iL[9]; inv3(R, iL);
  #pragma unroll
  for (int k = 0; k < 9; ++k) p[12 + k] = iL[k];
  const float bx=M[3], by=M[7], bz=M[11];     // t of [R|t]; inv -> -R^-1 t
  p[21] = -(iL[0]*bx + iL[1]*by + iL[2]*bz);
  p[22] = -(iL[3]*bx + iL[4]*by + iL[5]*bz);
  p[23] = -(iL[6]*bx + iL[7]*by + iL[8]*bz);
}

// =====================================================================
// Kernel 3: softmax over D + geometry -> depth probs, flat indices, counts
// one thread per (bn,hw)
// =====================================================================
__global__ void k_geom(const float* __restrict__ Y, const float* __restrict__ P,
                       float* __restrict__ depthp, int* __restrict__ flatTab,
                       float* __restrict__ cnt)
{
  const int t = blockIdx.x * blockDim.x + threadIdx.x;
  if (t >= BN_ * HW_) return;
  const int bn = t / HW_, hw = t % HW_;
  const float u = ((hw % WF_) + 0.5f) * 8.0f;   // W_IN/WF = 8
  const float v = ((hw / WF_) + 0.5f) * 8.0f;   // H_IN/HF = 8
  const float* p = P + bn * 24;
  const float ux = u - p[9], uy = v - p[10], uz = 1.0f - p[11];
  const float rx = p[0]*ux + p[1]*uy + p[2]*uz;
  const float ry = p[3]*ux + p[4]*uy + p[5]*uz;
  const float rz = p[6]*ux + p[7]*uy + p[8]*uz;

  float lg[D_]; float mx = -INFINITY;
  #pragma unroll
  for (int d = 0; d < D_; ++d) {
    lg[d] = Y[((size_t)bn * MOUT_ + d) * HW_ + hw];
    mx = fmaxf(mx, lg[d]);
  }
  float s = 0.0f;
  #pragma unroll
  for (int d = 0; d < D_; ++d) { lg[d] = __expf(lg[d] - mx); s += lg[d]; }
  const float is = 1.0f / s;
  const int bb = bn / 6;

  #pragma unroll
  for (int d = 0; d < D_; ++d) {
    depthp[(size_t)t * D_ + d] = lg[d] * is;
    const float dv = (d + 0.5f) * 2.8125f;              // (45/16)
    const float px = rx * dv, py = ry * dv, pz = rz * dv;
    const float X  = p[12]*px + p[13]*py + p[14]*pz + p[21];
    const float Yw = p[15]*px + p[16]*py + p[17]*pz + p[22];
    const float Z  = p[18]*px + p[19]*py + p[20]*pz + p[23];
    const int ix = (int)floorf((X  + 51.2f) * 1.25f);
    const int iy = (int)floorf((Yw + 51.2f) * 1.25f);
    const bool ok = (ix >= 0) && (ix < BEV_) && (iy >= 0) && (iy < BEV_) &&
                    (Z >= -5.0f) && (Z < 3.0f);
    const int flat = ok ? (bb * (BEV_*BEV_) + iy * BEV_ + ix) : -1;
    flatTab[(size_t)t * D_ + d] = flat;
    if (ok) atomicAdd(&cnt[flat], 1.0f);
  }
}

// =====================================================================
// Kernel 4: splat (scatter-add depth_prob * ctx into BEV accumulator)
// thread = (channel-group g of 16, bn, hw)
// =====================================================================
__global__ void k_splat(const float* __restrict__ Y, const float* __restrict__ depthp,
                        const int* __restrict__ flatTab, float* __restrict__ acc)
{
  const int t = blockIdx.x * blockDim.x + threadIdx.x;
  if (t >= 5 * BN_ * HW_) return;
  const int g  = t / (BN_ * HW_);
  const int q  = t % (BN_ * HW_);
  const int bn = q / HW_, hw = q % HW_;
  float cv[16];
  #pragma unroll
  for (int c = 0; c < 16; ++c)
    cv[c] = Y[((size_t)bn * MOUT_ + D_ + g * 16 + c) * HW_ + hw];
  const int*   ft = flatTab + (size_t)q * D_;
  const float* dp = depthp  + (size_t)q * D_;
  for (int d = 0; d < D_; ++d) {
    const int f = ft[d];
    if (f >= 0) {
      const float pr = dp[d];
      float* a = acc + (size_t)f * COUT_ + g * 16;
      #pragma unroll
      for (int c = 0; c < 16; ++c) atomicAdd(&a[c], pr * cv[c]);
    }
  }
}

// =====================================================================
// Kernel 5: normalize + repack BEV to padded NHWC bf16 (halo + chan pad)
// =====================================================================
__global__ void k_bev(const float* __restrict__ acc, const float* __restrict__ cnt,
                      __bf16* __restrict__ bev)
{
  const int t = blockIdx.x * blockDim.x + threadIdx.x;
  if (t >= B_ * BEV_ * BEV_ * COUT_) return;
  const int c = t % COUT_;
  int q = t / COUT_;
  const int x = q % BEV_; q /= BEV_;
  const int y = q % BEV_;
  const int b = q / BEV_;
  const int cell = (b * BEV_ + y) * BEV_ + x;
  const float v = acc[(size_t)cell * COUT_ + c] / fmaxf(cnt[cell], 1.0f);
  bev[(((size_t)(b * BPAD_ + y + 1)) * BPAD_ + (x + 1)) * CPAD_ + c] = (__bf16)v;
}

// =====================================================================
// Kernel 6: weight repack  (80,80,3,3) f32 -> (tap, o, i_pad96) bf16
// =====================================================================
__global__ void k_wprep(const float* __restrict__ w_ref, __bf16* __restrict__ wtap)
{
  const int t = blockIdx.x * blockDim.x + threadIdx.x;
  if (t >= 9 * COUT_ * COUT_) return;
  const int i = t % COUT_;
  int q = t / COUT_;
  const int o = q % COUT_;
  const int tap = q / COUT_;
  wtap[((size_t)tap * COUT_ + o) * CPAD_ + i] = (__bf16)w_ref[(o * COUT_ + i) * 9 + tap];
}

// =====================================================================
// Kernel 7: 3x3 conv as implicit GEMM (9 taps x 3 K-chunks of 32, bf16 WMMA)
// block = 160 threads (5 waves = 5 out-channel tiles); grid = B*128*8 pixel tiles
// NHWC bf16 BEV -> each B-lane column load is one contiguous 32B read
// =====================================================================
__global__ __launch_bounds__(160) void k_conv(
    const __bf16* __restrict__ bev, const __bf16* __restrict__ wtap,
    const float* __restrict__ b_ref, float* __restrict__ out)
{
  const int blk = blockIdx.x;
  const int x0 = (blk % 8) * 16;
  int q = blk / 8;
  const int y = q % BEV_;
  const int b = q / BEV_;
  const int lane = threadIdx.x & 31;
  const int wave = threadIdx.x >> 5;     // out-channel tile 0..4
  const int hn = lane & 15, half = lane >> 4;

  v8f c = {};
  #pragma unroll
  for (int tap = 0; tap < 9; ++tap) {
    const int dy = tap / 3 - 1, dx = tap % 3 - 1;
    const __bf16* brow = bev +
        (((size_t)(b * BPAD_ + y + 1 + dy)) * BPAD_ + (x0 + 1 + dx + hn)) * CPAD_ + half * 16;
    const __bf16* arow = wtap + ((size_t)tap * COUT_ + wave * 16 + hn) * CPAD_ + half * 16;
    #pragma unroll
    for (int ch = 0; ch < 3; ++ch) {
      const int ic0 = ch * 32;
      v16bf a, bb;
      #pragma unroll
      for (int j = 0; j < 8; ++j) {
        a[j]     = arow[ic0 + j];
        a[j + 8] = arow[ic0 + 8 + j];
      }
      #pragma unroll
      for (int j = 0; j < 16; ++j) bb[j] = brow[ic0 + j];
      c = __builtin_amdgcn_wmma_f32_16x16x32_bf16(false, a, false, bb,
                                                  (short)0, c, false, false);
    }
  }
  const int x = x0 + hn;
  #pragma unroll
  for (int v = 0; v < 8; ++v) {
    const int o = wave * 16 + v + half * 8;
    out[(((size_t)(b * COUT_ + o)) * BEV_ + y) * BEV_ + x] = c[v] + b_ref[o];
  }
}

// =====================================================================
extern "C" void kernel_launch(void* const* d_in, const int* in_sizes, int n_in,
                              void* d_out, int out_size, void* d_ws, size_t ws_size,
                              hipStream_t stream) {
  const float* x     = (const float*)d_in[0];
  const float* l2i   = (const float*)d_in[1];
  const float* aug   = (const float*)d_in[2];
  const float* w_dep = (const float*)d_in[3];
  const float* b_dep = (const float*)d_in[4];
  const float* w_ref = (const float*)d_in[5];
  const float* b_ref = (const float*)d_in[6];
  float* out = (float*)d_out;

  char* ws = (char*)d_ws;
  float*  Y      = (float*)(ws + OFF_Y);
  float*  depthp = (float*)(ws + OFF_DEPTH);
  int*    flatT  = (int*)  (ws + OFF_FLAT);
  float*  P      = (float*)(ws + OFF_PARAMS);
  float*  cnt    = (float*)(ws + OFF_CNT);
  float*  acc    = (float*)(ws + OFF_ACC);
  __bf16* bev    = (__bf16*)(ws + OFF_BEV);
  __bf16* wtap   = (__bf16*)(ws + OFF_WTAP);

  // zero the accumulator / padded-bf16 region once per launch (graph-safe)
  hipMemsetAsync(ws + OFF_ZERO, 0, OFF_END - OFF_ZERO, stream);

  k_params<<<1, 32, 0, stream>>>(l2i, aug, P);
  k_depth_gemm<<<BN_ * NB_, 192, 0, stream>>>(x, w_dep, b_dep, Y);
  {
    const int n = BN_ * HW_;
    k_geom<<<(n + 255) / 256, 256, 0, stream>>>(Y, P, depthp, flatT, cnt);
  }
  {
    const int n = 5 * BN_ * HW_;
    k_splat<<<(n + 255) / 256, 256, 0, stream>>>(Y, depthp, flatT, acc);
  }
  {
    const int n = B_ * BEV_ * BEV_ * COUT_;
    k_bev<<<(n + 255) / 256, 256, 0, stream>>>(acc, cnt, bev);
  }
  {
    const int n = 9 * COUT_ * COUT_;
    k_wprep<<<(n + 255) / 256, 256, 0, stream>>>(w_ref, wtap);
  }
  k_conv<<<B_ * BEV_ * 8, 160, 0, stream>>>(bev, wtap, b_ref, out);
}
